// LocationAwareMSAGAT_Net_30081950941206
// MI455X (gfx1250) — compile-verified, matrix operand order checked by reading, not run on hardware
//
#include <hip/hip_runtime.h>
#include <hip/hip_bf16.h>

#define BB    8
#define NN    1024
#define HH    256
#define SS    4
#define KK    3
#define HEADS 4
#define DD    64
#define PADR  8
#define NP    (NN + 2*PADR)

typedef __attribute__((ext_vector_type(16))) __bf16 v16bf;
typedef __attribute__((ext_vector_type(8)))  __bf16 v8bf;
typedef __attribute__((ext_vector_type(8)))  float  v8f;

// CDNA5 async global->LDS path (ASYNCcnt), with portable fallback.
#if defined(__AMDGCN__) && __has_builtin(__builtin_amdgcn_global_load_async_to_lds_b128)
#define USE_ASYNC 1
typedef int v4i_vs __attribute__((vector_size(16)));
typedef __attribute__((address_space(1))) v4i_vs* g_v4i_ptr;
typedef __attribute__((address_space(3))) v4i_vs* l_v4i_ptr;
#else
#define USE_ASYNC 0
#endif

__device__ __forceinline__ void copy16_g2l(const __bf16* g, __bf16* l) {
#if USE_ASYNC
    __builtin_amdgcn_global_load_async_to_lds_b128((g_v4i_ptr)g, (l_v4i_ptr)l, 0, 0);
#else
    *(v8bf*)l = *(const v8bf*)g;
#endif
}

__device__ __forceinline__ void async_join() {
#if USE_ASYNC
#if __has_builtin(__builtin_amdgcn_s_wait_asynccnt)
    __builtin_amdgcn_s_wait_asynccnt(0);
#else
    asm volatile("s_wait_asynccnt 0x0" ::: "memory");
#endif
#endif
}

__device__ __forceinline__ __bf16 f2bf(float f) {
    union { float f; unsigned u; } v; v.f = f;
    unsigned r = v.u + 0x7FFFu + ((v.u >> 16) & 1u);
    unsigned short h = (unsigned short)(r >> 16);
    return __builtin_bit_cast(__bf16, h);
}

__device__ __forceinline__ v16bf combine16(v8bf lo, v8bf hi) {
    return __builtin_shufflevector(lo, hi, 0,1,2,3,4,5,6,7,8,9,10,11,12,13,14,15);
}

// ---------------- prep kernels ----------------

__global__ __launch_bounds__(256) void k_pack_x(const float* __restrict__ x,
                                                __bf16* __restrict__ xb) {
    size_t idx = (size_t)blockIdx.x * 256 + threadIdx.x;
    size_t total = (size_t)BB * NP * HH;
    if (idx >= total) return;
    int c = idx % HH;
    int p = (idx / HH) % NP;
    int b = idx / ((size_t)NP * HH);
    int n = p - PADR;
    float v = (n >= 0 && n < NN) ? x[((size_t)b * NN + n) * HH + c] : 0.0f;
    xb[idx] = f2bf(v);
}

// conv_w [S,Ho,Hi,K] f32 -> Wt2 [(s*K+k)][o][i] bf16  (o-major, i inner)
__global__ __launch_bounds__(256) void k_pack_w(const float* __restrict__ cw,
                                                __bf16* __restrict__ Wt2) {
    size_t idx = (size_t)blockIdx.x * 256 + threadIdx.x;
    size_t total = (size_t)SS * KK * HH * HH;
    if (idx >= total) return;
    int i = idx % HH;
    int o = (idx / HH) % HH;
    int sk = idx / ((size_t)HH * HH);
    int s = sk / KK, k = sk % KK;
    float v = cw[(((size_t)s * HH + o) * HH + i) * KK + k];
    Wt2[idx] = f2bf(v);
}

// gat_W [heads,H,d] f32 -> Wg2 [o=head*64+dd][i] bf16
__global__ __launch_bounds__(256) void k_pack_g(const float* __restrict__ gw,
                                                __bf16* __restrict__ Wg2) {
    size_t idx = (size_t)blockIdx.x * 256 + threadIdx.x;
    size_t total = (size_t)HH * HH;
    if (idx >= total) return;
    int i = idx % HH;
    int o = idx / HH;
    int head = o >> 6, dd = o & 63;
    float v = gw[((size_t)head * HH + i) * DD + dd];
    Wg2[idx] = f2bf(v);
}

// ---------------- stage 1: multiscale conv + fusion + bottleneck + LN1 ----------------

__global__ __launch_bounds__(256)
void k_msconv(const __bf16* __restrict__ xb, const __bf16* __restrict__ Wt2,
              const float* __restrict__ x,
              const float* __restrict__ conv_b, const float* __restrict__ bn_g,
              const float* __restrict__ bn_b,   const float* __restrict__ fusion_w,
              const float* __restrict__ W_low,  const float* __restrict__ b_low,
              const float* __restrict__ W_high, const float* __restrict__ b_high,
              const float* __restrict__ ln1_g,  const float* __restrict__ ln1_b,
              __bf16* __restrict__ hb) {
    __shared__ union {
        struct { __align__(16) __bf16 X[48 * HH]; __align__(16) __bf16 Bm[2][HH * 32]; } s1;
        struct { float F[32 * HH]; float T[32 * 8]; float R1[256]; float R2[256];
                 float M[32]; float V[32]; } s2;
    } sh;
    const int tid  = threadIdx.x;
    const int wave = tid >> 5, lane = tid & 31;
    const int wm = wave & 1, wn = wave >> 1;
    const int m0 = blockIdx.x * 32;
    const int b  = m0 >> 10;
    const int n0 = m0 & (NN - 1);

    // stage X rows [n0-8, n0+40) padded -> 48 x 256 bf16 (direct global->LDS)
    {
        const __bf16* src = xb + (size_t)b * NP * HH + (size_t)n0 * HH;
        #pragma unroll
        for (int c = 0; c < 6; ++c) {
            int idx = tid + c * 256;
            int row = idx >> 5;
            int col8 = (idx & 31) * 8;
            copy16_g2l(&src[row * HH + col8], &sh.s1.X[row * HH + col8]);
        }
    }

    float alpha[SS];
    {
        float f0 = fusion_w[0], f1 = fusion_w[1], f2 = fusion_w[2], f3 = fusion_w[3];
        float mx = fmaxf(fmaxf(f0, f1), fmaxf(f2, f3));
        float e0 = __expf(f0 - mx), e1 = __expf(f1 - mx),
              e2 = __expf(f2 - mx), e3 = __expf(f3 - mx);
        float inv = 1.0f / (e0 + e1 + e2 + e3);
        alpha[0] = e0 * inv; alpha[1] = e1 * inv; alpha[2] = e2 * inv; alpha[3] = e3 * inv;
    }
    const float bnscale = rsqrtf(1.0f + 1e-5f);

    float fused[4][8];
    #pragma unroll
    for (int t = 0; t < 4; ++t)
        #pragma unroll
        for (int r = 0; r < 8; ++r) fused[t][r] = 0.0f;

    // weight slice stager: 32 K-elems of (s,k) for output column o=tid
    auto stage_w = [&](int s, int kt, int k, int buf) {
        const __bf16* wsrc = Wt2 + ((size_t)((s * KK + k) * HH + tid)) * HH + kt * 32;
        __bf16* d = &sh.s1.Bm[buf][tid * 32];
        copy16_g2l(wsrc + 0,  d + 0);
        copy16_g2l(wsrc + 8,  d + 8);
        copy16_g2l(wsrc + 16, d + 16);
        copy16_g2l(wsrc + 24, d + 24);
    };

    for (int s = 0; s < SS; ++s) {
        v8f acc[4];
        #pragma unroll
        for (int t = 0; t < 4; ++t)
            #pragma unroll
            for (int r = 0; r < 8; ++r) acc[t][r] = 0.0f;

        stage_w(s, 0, 0, 0);         // prologue: first slice into buffer 0
        async_join();
        __syncthreads();             // X + first slice visible

        for (int itr = 0; itr < 24; ++itr) {     // itr = kt*3 + k
            const int kt = itr / 3, k = itr % 3;
            const int buf = itr & 1;
            if (itr < 23) {                      // stage next slice into other buffer
                int ni = itr + 1;
                stage_w(s, ni / 3, ni % 3, buf ^ 1);
            }
            const int off   = (k - 1) << s;      // tap shift * dilation 2^s
            const int arow  = 8 + wm * 16 + (lane & 15) + off;
            const int kbase = kt * 32 + (lane >> 4) * 8;
            v8bf alo = *(const v8bf*)&sh.s1.X[arow * HH + kbase];
            v8bf ahi = *(const v8bf*)&sh.s1.X[arow * HH + kbase + 16];
            v16bf af = combine16(alo, ahi);
            const int kb = (lane >> 4) * 8;
            #pragma unroll
            for (int t = 0; t < 4; ++t) {
                int o = wn * 64 + t * 16 + (lane & 15);
                v8bf blo = *(const v8bf*)&sh.s1.Bm[buf][o * 32 + kb];
                v8bf bhi = *(const v8bf*)&sh.s1.Bm[buf][o * 32 + kb + 16];
                v16bf bf = combine16(blo, bhi);
                acc[t] = __builtin_amdgcn_wmma_f32_16x16x32_bf16(
                    false, af, false, bf, (short)0, acc[t], false, false);
            }
            async_join();
            __syncthreads();
        }

        // per-scale epilogue: bias, eval-BN, SiLU, weighted fusion
        #pragma unroll
        for (int t = 0; t < 4; ++t)
            #pragma unroll
            for (int r = 0; r < 8; ++r) {
                int col = wn * 64 + t * 16 + (lane & 15);
                float v = acc[t][r] + conv_b[s * HH + col];
                v = v * bnscale * bn_g[s * HH + col] + bn_b[s * HH + col];
                v = v / (1.0f + __expf(-v));   // SiLU
                fused[t][r] += alpha[s] * v;
            }
    }

    __syncthreads();
    // scatter fused tile to LDS f32 [32][256]
    #pragma unroll
    for (int t = 0; t < 4; ++t)
        #pragma unroll
        for (int r = 0; r < 8; ++r) {
            int rloc = wm * 16 + r + 8 * (lane >> 4);
            int col  = wn * 64 + t * 16 + (lane & 15);
            sh.s2.F[rloc * HH + col] = fused[t][r];
        }
    __syncthreads();

    // bottleneck low: 32 rows x 8 cols, one thread each
    {
        int row = tid >> 3, c = tid & 7;
        float a = b_low[c];
        for (int i = 0; i < HH; ++i) a += sh.s2.F[row * HH + i] * W_low[i * 8 + c];
        sh.s2.T[row * 8 + c] = a;
    }
    __syncthreads();

    // high + residual + LN1
    {
        int rloc = tid >> 3, part = tid & 7;
        float tl[8];
        #pragma unroll
        for (int c = 0; c < 8; ++c) tl[c] = sh.s2.T[rloc * 8 + c];
        float vals[32];
        float ps = 0.f, ps2 = 0.f;
        #pragma unroll
        for (int j = 0; j < 32; ++j) {
            int i = part * 32 + j;
            float v = b_high[i] + x[((size_t)(m0 + rloc)) * HH + i];
            #pragma unroll
            for (int c = 0; c < 8; ++c) v += tl[c] * W_high[c * HH + i];
            vals[j] = v; ps += v; ps2 += v * v;
        }
        sh.s2.R1[tid] = ps; sh.s2.R2[tid] = ps2;
        __syncthreads();
        if (part == 0) {
            float a = 0.f, q = 0.f;
            #pragma unroll
            for (int p = 0; p < 8; ++p) { a += sh.s2.R1[rloc * 8 + p]; q += sh.s2.R2[rloc * 8 + p]; }
            float mean = a / 256.0f;
            float var  = q / 256.0f - mean * mean;
            sh.s2.M[rloc] = mean;
            sh.s2.V[rloc] = rsqrtf(var + 1e-5f);
        }
        __syncthreads();
        float mean = sh.s2.M[rloc], rstd = sh.s2.V[rloc];
        #pragma unroll
        for (int j = 0; j < 32; ++j) {
            int i = part * 32 + j;
            float hv = (vals[j] - mean) * rstd * ln1_g[i] + ln1_b[i];
            hb[((size_t)(m0 + rloc)) * HH + i] = f2bf(hv);
        }
    }
}

// ---------------- stage 2: GAT projection + src/dst ----------------

__global__ __launch_bounds__(256)
void k_gatproj(const __bf16* __restrict__ hb, const __bf16* __restrict__ Wg2,
               const float* __restrict__ a_src, const float* __restrict__ a_dst,
               __bf16* __restrict__ WhT, float* __restrict__ srcv, float* __restrict__ dstv) {
    __shared__ union {
        struct { __align__(16) __bf16 A[32 * HH]; __align__(16) __bf16 Bm[2][HH * 32]; } s1;
        struct { float F[32 * HH]; } s2;
    } sh;
    const int tid  = threadIdx.x;
    const int wave = tid >> 5, lane = tid & 31;
    const int wm = wave & 1, wn = wave >> 1;
    const int m0 = blockIdx.x * 32;
    const int b  = m0 >> 10;

    // stage A: 32 x 256 bf16 (direct global->LDS)
    #pragma unroll
    for (int c = 0; c < 4; ++c) {
        int idx = tid + c * 256;
        int row = idx >> 5;
        int col8 = (idx & 31) * 8;
        copy16_g2l(&hb[((size_t)(m0 + row)) * HH + col8], &sh.s1.A[row * HH + col8]);
    }

    auto stage_w = [&](int kt, int buf) {
        const __bf16* wsrc = Wg2 + (size_t)tid * HH + kt * 32;
        __bf16* d = &sh.s1.Bm[buf][tid * 32];
        copy16_g2l(wsrc + 0,  d + 0);
        copy16_g2l(wsrc + 8,  d + 8);
        copy16_g2l(wsrc + 16, d + 16);
        copy16_g2l(wsrc + 24, d + 24);
    };

    v8f acc[4];
    #pragma unroll
    for (int t = 0; t < 4; ++t)
        #pragma unroll
        for (int r = 0; r < 8; ++r) acc[t][r] = 0.0f;

    stage_w(0, 0);
    async_join();
    __syncthreads();

    for (int kt = 0; kt < 8; ++kt) {
        const int buf = kt & 1;
        if (kt < 7) stage_w(kt + 1, buf ^ 1);
        const int arow  = wm * 16 + (lane & 15);
        const int kbase = kt * 32 + (lane >> 4) * 8;
        v8bf alo = *(const v8bf*)&sh.s1.A[arow * HH + kbase];
        v8bf ahi = *(const v8bf*)&sh.s1.A[arow * HH + kbase + 16];
        v16bf af = combine16(alo, ahi);
        const int kb = (lane >> 4) * 8;
        #pragma unroll
        for (int t = 0; t < 4; ++t) {
            int o = wn * 64 + t * 16 + (lane & 15);
            v8bf blo = *(const v8bf*)&sh.s1.Bm[buf][o * 32 + kb];
            v8bf bhi = *(const v8bf*)&sh.s1.Bm[buf][o * 32 + kb + 16];
            v16bf bf = combine16(blo, bhi);
            acc[t] = __builtin_amdgcn_wmma_f32_16x16x32_bf16(
                false, af, false, bf, (short)0, acc[t], false, false);
        }
        async_join();
        __syncthreads();
    }

    // epilogue: Wh tile -> LDS f32 + transposed bf16 store for attention V
    #pragma unroll
    for (int t = 0; t < 4; ++t)
        #pragma unroll
        for (int r = 0; r < 8; ++r) {
            int rloc = wm * 16 + r + 8 * (lane >> 4);
            int col  = wn * 64 + t * 16 + (lane & 15);
            float v = acc[t][r];
            sh.s2.F[rloc * HH + col] = v;
            int head = col >> 6, dd = col & 63;
            int n = (m0 & (NN - 1)) + rloc;
            WhT[(((size_t)b * HEADS + head) * DD + dd) * NN + n] = f2bf(v);
        }
    __syncthreads();

    if (tid < 128) {
        int rloc = tid >> 2, head = tid & 3;
        float ssum = 0.f, dsum = 0.f;
        for (int d = 0; d < DD; ++d) {
            float w = sh.s2.F[rloc * HH + head * DD + d];
            ssum += w * a_src[head * DD + d];
            dsum += w * a_dst[head * DD + d];
        }
        int n = (m0 & (NN - 1)) + rloc;
        srcv[((size_t)b * HEADS + head) * NN + n] = ssum;
        dstv[((size_t)b * HEADS + head) * NN + n] = dsum;
    }
}

// ---------------- stage 3: masked attention (flash-style, P never hits HBM) ----------------

__global__ __launch_bounds__(256)
void k_attn(const float* __restrict__ srcv, const float* __restrict__ dstv,
            const int* __restrict__ adj, const __bf16* __restrict__ WhT,
            float* __restrict__ hp) {
    __shared__ struct {
        __align__(16) __bf16 P[16 * NN];   // exp(scores - rowmax), unnormalized
        float O[16 * DD];
        float red[16 * 16];
        float rsum[16];
    } sh;
    const int tid  = threadIdx.x;
    const int wave = tid >> 5, lane = tid & 31;
    const int wg = blockIdx.x;
    const int b = wg >> 8;
    const int head = (wg >> 6) & 3;
    const int m0 = (wg & 63) * 16;

    const float* srow = srcv + ((size_t)b * HEADS + head) * NN;
    const float* drow = dstv + ((size_t)b * HEADS + head) * NN;

    // phase 1: scores + rowwise softmax numerators
    const int r = tid >> 4, cpart = tid & 15;
    const float sm = srow[m0 + r];
    const int* arow = adj + (size_t)(m0 + r) * NN;
    float scores[64];
    float mymax = -3.0e38f;
    #pragma unroll 8
    for (int j = 0; j < 64; ++j) {
        int n = cpart * 64 + j;
        float sc = sm + drow[n];
        sc = sc >= 0.0f ? sc : 0.2f * sc;      // leaky relu
        sc = arow[n] > 0 ? sc : -1.0e9f;       // adjacency mask
        scores[j] = sc;
        mymax = fmaxf(mymax, sc);
    }
    sh.red[r * 16 + cpart] = mymax;
    __syncthreads();
    float rowmax = sh.red[r * 16];
    #pragma unroll
    for (int p = 1; p < 16; ++p) rowmax = fmaxf(rowmax, sh.red[r * 16 + p]);
    __syncthreads();
    float psum = 0.f;
    #pragma unroll 8
    for (int j = 0; j < 64; ++j) {
        float e = __expf(scores[j] - rowmax);
        psum += e;
        sh.P[r * NN + cpart * 64 + j] = f2bf(e);
    }
    sh.red[r * 16 + cpart] = psum;
    __syncthreads();
    float tsum = 0.f;
    #pragma unroll
    for (int p = 0; p < 16; ++p) tsum += sh.red[r * 16 + p];
    if (cpart == 0) sh.rsum[r] = 1.0f / tsum;
    for (int i = tid; i < 16 * DD; i += 256) sh.O[i] = 0.0f;
    __syncthreads();

    // phase 2: O[16x64] = P[16x1024] @ V, K split across 8 waves
    const __bf16* vbase = WhT + ((size_t)b * HEADS + head) * DD * NN;
    v8f acc[4];
    #pragma unroll
    for (int t = 0; t < 4; ++t)
        #pragma unroll
        for (int rr = 0; rr < 8; ++rr) acc[t][rr] = 0.0f;
    #pragma unroll
    for (int q = 0; q < 4; ++q) {
        int kc = wave * 4 + q;
        int prow = lane & 15;
        int kb = kc * 32 + (lane >> 4) * 8;
        v8bf alo = *(const v8bf*)&sh.P[prow * NN + kb];
        v8bf ahi = *(const v8bf*)&sh.P[prow * NN + kb + 16];
        v16bf af = combine16(alo, ahi);
        #pragma unroll
        for (int t = 0; t < 4; ++t) {
            int dd = t * 16 + (lane & 15);
            const __bf16* vp = vbase + (size_t)dd * NN + kb;
            v8bf blo = *(const v8bf*)vp;
            v8bf bhi = *(const v8bf*)(vp + 16);
            v16bf bf = combine16(blo, bhi);
            acc[t] = __builtin_amdgcn_wmma_f32_16x16x32_bf16(
                false, af, false, bf, (short)0, acc[t], false, false);
        }
    }
    #pragma unroll
    for (int t = 0; t < 4; ++t)
        #pragma unroll
        for (int rr = 0; rr < 8; ++rr) {
            int rloc = rr + 8 * (lane >> 4);
            int dd = t * 16 + (lane & 15);
            atomicAdd(&sh.O[rloc * DD + dd], acc[t][rr]);
        }
    __syncthreads();
    for (int i = tid; i < 16 * DD; i += 256) {
        int rr = i >> 6, dd = i & 63;
        float v = sh.O[rr * DD + dd] * sh.rsum[rr];
        hp[((size_t)b * NN + m0 + rr) * HH + head * DD + dd] = v;
    }
}

// ---------------- stage 4: final layer norm ----------------

__global__ __launch_bounds__(256)
void k_ln2(const float* __restrict__ hp, const float* __restrict__ g,
           const float* __restrict__ bta, float* __restrict__ out) {
    __shared__ float red1[8][32], red2[8][32], mv[8][2];
    const int tid = threadIdx.x;
    const int r = tid >> 5, lane = tid & 31;
    const size_t row = (size_t)blockIdx.x * 8 + r;
    const float* p = hp + row * HH;
    float v[8];
    float ps = 0.f, ps2 = 0.f;
    #pragma unroll
    for (int j = 0; j < 8; ++j) {
        v[j] = p[lane + j * 32];
        ps += v[j]; ps2 += v[j] * v[j];
    }
    red1[r][lane] = ps; red2[r][lane] = ps2;
    __syncthreads();
    if (lane == 0) {
        float a = 0.f, q = 0.f;
        for (int i = 0; i < 32; ++i) { a += red1[r][i]; q += red2[r][i]; }
        float mean = a / 256.0f;
        float var  = q / 256.0f - mean * mean;
        mv[r][0] = mean; mv[r][1] = rsqrtf(var + 1e-5f);
    }
    __syncthreads();
    float mean = mv[r][0], rstd = mv[r][1];
    float* o = out + row * HH;
    #pragma unroll
    for (int j = 0; j < 8; ++j) {
        int i = lane + j * 32;
        o[i] = (v[j] - mean) * rstd * g[i] + bta[i];
    }
}

// ---------------- host launcher ----------------

extern "C" void kernel_launch(void* const* d_in, const int* in_sizes, int n_in,
                              void* d_out, int out_size, void* d_ws, size_t ws_size,
                              hipStream_t stream) {
    const float* x        = (const float*)d_in[0];
    const int*   adj      = (const int*)  d_in[1];
    const float* conv_w   = (const float*)d_in[2];
    const float* conv_b   = (const float*)d_in[3];
    const float* bn_g     = (const float*)d_in[4];
    const float* bn_b     = (const float*)d_in[5];
    const float* fusion_w = (const float*)d_in[6];
    const float* W_low    = (const float*)d_in[7];
    const float* b_low    = (const float*)d_in[8];
    const float* W_high   = (const float*)d_in[9];
    const float* b_high   = (const float*)d_in[10];
    const float* ln1_g    = (const float*)d_in[11];
    const float* ln1_b    = (const float*)d_in[12];
    const float* gat_W    = (const float*)d_in[13];
    const float* a_src    = (const float*)d_in[14];
    const float* a_dst    = (const float*)d_in[15];
    const float* ln2_g    = (const float*)d_in[16];
    const float* ln2_b    = (const float*)d_in[17];
    float* out = (float*)d_out;

    char* ws = (char*)d_ws;
    size_t off = 0;
    auto alloc = [&](size_t bytes) -> void* {
        off = (off + 255) & ~(size_t)255;
        void* p = ws + off;
        off += bytes;
        return p;
    };
    __bf16* xb   = (__bf16*)alloc((size_t)BB * NP * HH * 2);
    __bf16* Wt2  = (__bf16*)alloc((size_t)SS * KK * HH * HH * 2);
    __bf16* Wg2  = (__bf16*)alloc((size_t)HH * HH * 2);
    __bf16* hb   = (__bf16*)alloc((size_t)BB * NN * HH * 2);
    __bf16* WhT  = (__bf16*)alloc((size_t)BB * NN * HH * 2);
    float*  srcv = (float*) alloc((size_t)BB * HEADS * NN * 4);
    float*  dstv = (float*) alloc((size_t)BB * HEADS * NN * 4);
    float*  hp   = (float*) alloc((size_t)BB * NN * HH * 4);
    (void)ws_size; (void)in_sizes; (void)n_in; (void)out_size;

    {
        size_t total = (size_t)BB * NP * HH;
        k_pack_x<<<dim3((total + 255) / 256), dim3(256), 0, stream>>>(x, xb);
    }
    {
        size_t total = (size_t)SS * KK * HH * HH;
        k_pack_w<<<dim3((total + 255) / 256), dim3(256), 0, stream>>>(conv_w, Wt2);
    }
    {
        size_t total = (size_t)HH * HH;
        k_pack_g<<<dim3((total + 255) / 256), dim3(256), 0, stream>>>(gat_W, Wg2);
    }

    k_msconv<<<dim3(BB * NN / 32), dim3(256), 0, stream>>>(
        xb, Wt2, x, conv_b, bn_g, bn_b, fusion_w,
        W_low, b_low, W_high, b_high, ln1_g, ln1_b, hb);

    k_gatproj<<<dim3(BB * NN / 32), dim3(256), 0, stream>>>(
        hb, Wg2, a_src, a_dst, WhT, srcv, dstv);

    k_attn<<<dim3(BB * HEADS * (NN / 16)), dim3(256), 0, stream>>>(
        srcv, dstv, adj, WhT, hp);

    k_ln2<<<dim3(BB * NN / 8), dim3(256), 0, stream>>>(hp, ln2_g, ln2_b, out);
}
